// Block_3908420240004
// MI455X (gfx1250) — compile-verified, matrix operand order checked by reading, not run on hardware
//
#include <hip/hip_runtime.h>
#include <math.h>

typedef __bf16 bf16_t;
typedef __attribute__((ext_vector_type(8)))  __bf16 v8bf;
typedef __attribute__((ext_vector_type(16))) __bf16 v16bf;
typedef __attribute__((ext_vector_type(8)))  float  v8f;
typedef __attribute__((ext_vector_type(4)))  int    v4i32;

#define DEV __device__ __forceinline__

// ------------------------- CDNA5 async global->LDS ------------------------
#if __has_builtin(__builtin_amdgcn_global_load_async_to_lds_b128)
#define USE_ASYNC_LDS 1
#endif

DEV void async_cp16(void* l, const void* g) {
#ifdef USE_ASYNC_LDS
    __builtin_amdgcn_global_load_async_to_lds_b128(
        (__attribute__((address_space(1))) v4i32*)(void*)g,
        (__attribute__((address_space(3))) v4i32*)l, 0, 0);
#else
    *(uint4*)l = *(const uint4*)g;
#endif
}

DEV void wait_async0() {
#ifdef USE_ASYNC_LDS
#if __has_builtin(__builtin_amdgcn_s_wait_asynccnt)
    __builtin_amdgcn_s_wait_asynccnt(0);
#else
    asm volatile("s_wait_asynccnt 0x0" ::: "memory");
#endif
#endif
}

DEV v16bf ld16(const bf16_t* lo, const bf16_t* hi) {
    v8bf a = *(const v8bf*)lo;
    v8bf b = *(const v8bf*)hi;
    return __builtin_shufflevector(a, b, 0,1,2,3,4,5,6,7,8,9,10,11,12,13,14,15);
}

DEV v8f wmma_bf16f32(v16bf a, v16bf b, v8f c) {
    // D = A(16x32 bf16) * B(32x16 bf16) + C(16x16 f32)
    return __builtin_amdgcn_wmma_f32_16x16x32_bf16(false, a, false, b, (short)0, c, false, false);
}

// ---------------------------------------------------------------------------
// Weight conversion kernels
// ---------------------------------------------------------------------------

// Pack Wq/Wk/Wv ([H,D,hs] f32 each) into bf16 B^T layout: out[n][d], n = w*D + h*64 + e
__global__ void cvt_wqkv_kernel(const float* __restrict__ Wq,
                                const float* __restrict__ Wk,
                                const float* __restrict__ Wv,
                                bf16_t* __restrict__ out, int D) {
    long idx = (long)blockIdx.x * 256 + threadIdx.x;
    long total = 3L * D * D;
    if (idx >= total) return;
    long d = idx % D;
    long n = idx / D;
    int  w = (int)(n / D);
    int  nn = (int)(n % D);
    int  hh = nn >> 6;          // hs = 64
    int  e  = nn & 63;
    const float* W = (w == 0) ? Wq : ((w == 1) ? Wk : Wv);
    out[idx] = (bf16_t)W[((long)hh * D + d) * 64 + e];
}

// out[n*K + k] = W[k*N + n]  (f32 [K,N] -> bf16 [N,K])
__global__ void transpose_cvt_kernel(const float* __restrict__ W,
                                     bf16_t* __restrict__ out, int K, int N) {
    long idx = (long)blockIdx.x * 256 + threadIdx.x;
    long total = (long)K * N;
    if (idx >= total) return;
    long k = idx % K;
    long n = idx / K;
    out[idx] = (bf16_t)W[k * (long)N + n];
}

// ---------------------------------------------------------------------------
// LayerNorm (optionally adds a second tensor first): one block per row, D=1024
// ---------------------------------------------------------------------------
__global__ __launch_bounds__(256)
void ln_kernel(const float* __restrict__ X, const float* __restrict__ Xadd,
               const float* __restrict__ g, const float* __restrict__ be,
               float* __restrict__ Yf, bf16_t* __restrict__ Yb, int D) {
    long row = blockIdx.x;
    const float* x = X + row * (long)D;
    const float* xa = Xadd ? (Xadd + row * (long)D) : nullptr;
    __shared__ float red[16];
    float v[4];
    float s = 0.f, sq = 0.f;
#pragma unroll
    for (int i = 0; i < 4; ++i) {
        int c = threadIdx.x + i * 256;
        float t = x[c];
        if (xa) t += xa[c];
        v[i] = t; s += t; sq += t * t;
    }
#pragma unroll
    for (int m = 1; m < 32; m <<= 1) {
        s  += __shfl_xor(s, m, 32);
        sq += __shfl_xor(sq, m, 32);
    }
    int wv = threadIdx.x >> 5;
    if ((threadIdx.x & 31) == 0) { red[wv] = s; red[8 + wv] = sq; }
    __syncthreads();
    s = 0.f; sq = 0.f;
#pragma unroll
    for (int i = 0; i < 8; ++i) { s += red[i]; sq += red[8 + i]; }
    float mean = s / (float)D;
    float var  = (sq - (float)D * mean * mean) / (float)(D - 1);  // ddof=1
    float rstd = rsqrtf(var + 1e-5f);
#pragma unroll
    for (int i = 0; i < 4; ++i) {
        int c = threadIdx.x + i * 256;
        float y = g[c] * ((v[i] - mean) * rstd) + be[c];
        if (Yf) Yf[row * (long)D + c] = y;
        if (Yb) Yb[row * (long)D + c] = (bf16_t)y;
    }
}

// ---------------------------------------------------------------------------
// bf16 WMMA GEMM: C[M,N] = A[M,K] * B, with B given transposed as BT[N,K].
// Block tile 64x256, 8 waves, each wave 32x64 (2x4 WMMA tiles), Kc = 32,
// double-buffered LDS with async global->LDS staging (one barrier / K-step).
// Epilogue: optional bias[N], ReLU, residual[M,N] (f32), f32 and/or bf16 out.
// ---------------------------------------------------------------------------
__global__ __launch_bounds__(256)
void gemm_bf16_kernel(const bf16_t* __restrict__ A,
                      const bf16_t* __restrict__ BT,
                      int M, int N, int K,
                      const float* __restrict__ bias, int relu,
                      const float* __restrict__ resid,
                      float* __restrict__ Cf, bf16_t* __restrict__ Cb) {
    (void)M;
    __shared__ __align__(16) bf16_t As[2][64][32];    //  8 KB
    __shared__ __align__(16) bf16_t Bs[2][256][32];   // 32 KB
    const int tid  = threadIdx.x;
    const int L    = tid & 31;
    const int wv   = tid >> 5;
    const int half = L >> 4;
    const int l16  = L & 15;
    const int wrow = wv & 1;        // 2 wave-rows of 32
    const int wcol = wv >> 1;       // 4 wave-cols of 64
    const long mbase = (long)blockIdx.y * 64;
    const long nbase = (long)blockIdx.x * 256;

    const int ar = tid >> 2, ac = (tid & 3) * 8;      // A cooperative load

    auto stage = [&](int it, int buf) {
        long k0 = (long)it * 32;
        async_cp16(&As[buf][ar][ac], &A[(mbase + ar) * (long)K + k0 + ac]);
        const bf16_t* bp = &BT[(nbase + tid) * (long)K + k0];
#pragma unroll
        for (int j = 0; j < 4; ++j)
            async_cp16(&Bs[buf][tid][j * 8], bp + j * 8);
    };

    const int niter = K >> 5;
    stage(0, 0);

    v8f c[2][4] = {};
    for (int it = 0; it < niter; ++it) {
        const int cur = it & 1;
        wait_async0();
        __syncthreads();
        if (it + 1 < niter) stage(it + 1, cur ^ 1);

        v16bf af[2], bf[4];
#pragma unroll
        for (int mt = 0; mt < 2; ++mt) {
            int r = wrow * 32 + mt * 16 + l16;
            af[mt] = ld16(&As[cur][r][half * 8], &As[cur][r][16 + half * 8]);
        }
#pragma unroll
        for (int nt = 0; nt < 4; ++nt) {
            int cc = wcol * 64 + nt * 16 + l16;
            bf[nt] = ld16(&Bs[cur][cc][half * 8], &Bs[cur][cc][16 + half * 8]);
        }
#pragma unroll
        for (int mt = 0; mt < 2; ++mt)
#pragma unroll
            for (int nt = 0; nt < 4; ++nt)
                c[mt][nt] = wmma_bf16f32(af[mt], bf[nt], c[mt][nt]);
    }
    // epilogue
#pragma unroll
    for (int mt = 0; mt < 2; ++mt) {
#pragma unroll
        for (int nt = 0; nt < 4; ++nt) {
#pragma unroll
            for (int i = 0; i < 8; ++i) {
                long r  = mbase + wrow * 32 + mt * 16 + half * 8 + i;
                long cc = nbase + wcol * 64 + nt * 16 + l16;
                float v = c[mt][nt][i];
                if (bias)  v += bias[cc];
                if (relu)  v = fmaxf(v, 0.f);
                if (resid) v += resid[r * (long)N + cc];
                if (Cf) Cf[r * (long)N + cc] = v;
                if (Cb) Cb[r * (long)N + cc] = (bf16_t)v;
            }
        }
    }
}

// ---------------------------------------------------------------------------
// QKV rearrange: tmp[BT,3D] bf16 -> Q/K/V [B,H,T,hs] bf16; Q pre-scaled.
// ---------------------------------------------------------------------------
__global__ void qkv_rearr_kernel(const bf16_t* __restrict__ tmp,
                                 const float* __restrict__ bq,
                                 const float* __restrict__ bk,
                                 const float* __restrict__ bv,
                                 bf16_t* __restrict__ Qb, bf16_t* __restrict__ Kb,
                                 bf16_t* __restrict__ Vb,
                                 int T, int H, float qscale) {
    long idx = (long)blockIdx.x * 256 + threadIdx.x;  // over B*H*T*64
    int  e  = (int)(idx & 63);
    long r  = idx >> 6;
    int  t  = (int)(r % T);
    long r2 = r / T;
    int  hh = (int)(r2 % H);
    int  b  = (int)(r2 / H);
    long bt = (long)b * T + t;
    long src = bt * 3072 + (long)hh * 64 + e;   // 3D = 3072, D = 1024
    float q  = (float)tmp[src]        + bq[hh * 64 + e];
    float k  = (float)tmp[src + 1024] + bk[hh * 64 + e];
    float vv = (float)tmp[src + 2048] + bv[hh * 64 + e];
    Qb[idx] = (bf16_t)(q * qscale);
    Kb[idx] = (bf16_t)k;
    Vb[idx] = (bf16_t)vv;
}

// ---------------------------------------------------------------------------
// Flash attention: grid (T/64, B*H), 128 threads = 4 waves, wave owns 16 rows.
// Q pre-scaled by D^-0.5. Causal. hs = 64.
// ---------------------------------------------------------------------------
__global__ __launch_bounds__(128)
void attn_kernel(const bf16_t* __restrict__ Q, const bf16_t* __restrict__ Km,
                 const bf16_t* __restrict__ V, float* __restrict__ O,
                 int T, int H, int Dm) {
    __shared__ __align__(16) bf16_t Ks[32][64];     // keys x hs
    __shared__ __align__(16) bf16_t VsT[64][32];    // hs x keys (transposed)
    __shared__ __align__(16) bf16_t Ps[4][16][32];  // per-wave P staging
    const int tid  = threadIdx.x;
    const int L    = tid & 31;
    const int wv   = tid >> 5;
    const int half = L >> 4;
    const int l16  = L & 15;
    const int bh   = blockIdx.y;
    const int b    = bh / H;
    const int hd   = bh % H;
    const long base = (long)bh * T * 64;
    const int qbase = blockIdx.x * 64 + wv * 16;

    v16bf qa[2];
    {
        const bf16_t* qp = Q + base + (long)(qbase + l16) * 64;
        qa[0] = ld16(qp + half * 8,      qp + 16 + half * 8);
        qa[1] = ld16(qp + 32 + half * 8, qp + 48 + half * 8);
    }
    v8f o[4] = {};
    float mrow[8], lrow[8];
#pragma unroll
    for (int i = 0; i < 8; ++i) { mrow[i] = -__builtin_inff(); lrow[i] = 0.f; }

    const int jend = blockIdx.x * 64 + 64;  // causal limit for this block
    const int kr = tid >> 2, kc = (tid & 3) * 16;
    for (int j0 = 0; j0 < jend; j0 += 32) {
        // stage K (async, row-major) and V (transposed, sync) into LDS
        const bf16_t* kp = Km + base + (long)(j0 + kr) * 64 + kc;
        async_cp16(&Ks[kr][kc], kp);
        async_cp16(&Ks[kr][kc + 8], kp + 8);
        const bf16_t* vp = V + base + (long)(j0 + kr) * 64 + kc;
        v8bf v0 = *(const v8bf*)vp;
        v8bf v1 = *(const v8bf*)(vp + 8);
#pragma unroll
        for (int i = 0; i < 8; ++i) VsT[kc + i][kr] = v0[i];
#pragma unroll
        for (int i = 0; i < 8; ++i) VsT[kc + 8 + i][kr] = v1[i];
        wait_async0();
        __syncthreads();

        // S = Q * K^T  (16 x 32 per wave)
        v8f s[2] = {};
#pragma unroll
        for (int nt = 0; nt < 2; ++nt) {
            int key = nt * 16 + l16;
            v16bf kb0 = ld16(&Ks[key][half * 8],      &Ks[key][16 + half * 8]);
            v16bf kb1 = ld16(&Ks[key][32 + half * 8], &Ks[key][48 + half * 8]);
            s[nt] = wmma_bf16f32(qa[0], kb0, s[nt]);
            s[nt] = wmma_bf16f32(qa[1], kb1, s[nt]);
        }
        // causal mask
#pragma unroll
        for (int nt = 0; nt < 2; ++nt)
#pragma unroll
            for (int i = 0; i < 8; ++i) {
                int qg = qbase + half * 8 + i;
                int kg = j0 + nt * 16 + l16;
                if (kg > qg) s[nt][i] = -__builtin_inff();
            }
        // online softmax: row stats via shfl within 16-lane halves
        float rmax[8];
#pragma unroll
        for (int i = 0; i < 8; ++i) rmax[i] = fmaxf(s[0][i], s[1][i]);
#pragma unroll
        for (int m = 1; m < 16; m <<= 1)
#pragma unroll
            for (int i = 0; i < 8; ++i) rmax[i] = fmaxf(rmax[i], __shfl_xor(rmax[i], m, 32));
        float alpha[8];
#pragma unroll
        for (int i = 0; i < 8; ++i) {
            float mn = fmaxf(mrow[i], rmax[i]);
            alpha[i] = __expf(mrow[i] - mn);
            mrow[i]  = mn;
        }
        float rsum[8];
#pragma unroll
        for (int i = 0; i < 8; ++i) rsum[i] = 0.f;
#pragma unroll
        for (int nt = 0; nt < 2; ++nt)
#pragma unroll
            for (int i = 0; i < 8; ++i) {
                float p = __expf(s[nt][i] - mrow[i]);
                s[nt][i] = p;
                rsum[i] += p;
            }
#pragma unroll
        for (int m = 1; m < 16; m <<= 1)
#pragma unroll
            for (int i = 0; i < 8; ++i) rsum[i] += __shfl_xor(rsum[i], m, 32);
#pragma unroll
        for (int i = 0; i < 8; ++i) lrow[i] = lrow[i] * alpha[i] + rsum[i];
#pragma unroll
        for (int t2 = 0; t2 < 4; ++t2)
#pragma unroll
            for (int i = 0; i < 8; ++i) o[t2][i] *= alpha[i];

        // P through wave-private LDS (same-wave DS ops are in-order)
#pragma unroll
        for (int nt = 0; nt < 2; ++nt)
#pragma unroll
            for (int i = 0; i < 8; ++i)
                Ps[wv][half * 8 + i][nt * 16 + l16] = (bf16_t)s[nt][i];
        v16bf pa = ld16(&Ps[wv][l16][half * 8], &Ps[wv][l16][16 + half * 8]);
        // O += P * V
#pragma unroll
        for (int nt2 = 0; nt2 < 4; ++nt2) {
            int col = nt2 * 16 + l16;
            v16bf vb = ld16(&VsT[col][half * 8], &VsT[col][16 + half * 8]);
            o[nt2] = wmma_bf16f32(pa, vb, o[nt2]);
        }
        __syncthreads();
    }
    // write O (f32), heads concatenated on D
#pragma unroll
    for (int i = 0; i < 8; ++i) {
        float inv = 1.f / lrow[i];
        int q = qbase + half * 8 + i;
#pragma unroll
        for (int nt2 = 0; nt2 < 4; ++nt2) {
            int c = nt2 * 16 + l16;
            O[((long)b * T + q) * (long)Dm + hd * 64 + c] = o[nt2][i] * inv;
        }
    }
}

// ---------------------------------------------------------------------------
// Host-side orchestration
// ---------------------------------------------------------------------------
extern "C" void kernel_launch(void* const* d_in, const int* in_sizes, int n_in,
                              void* d_out, int out_size, void* d_ws, size_t ws_size,
                              hipStream_t stream) {
    (void)in_sizes; (void)n_in; (void)out_size; (void)ws_size;
    const float* x   = (const float*)d_in[0];
    const float* Wq  = (const float*)d_in[1];
    const float* bq  = (const float*)d_in[2];
    const float* Wk  = (const float*)d_in[3];
    const float* bk  = (const float*)d_in[4];
    const float* Wv  = (const float*)d_in[5];
    const float* bv  = (const float*)d_in[6];
    const float* g1  = (const float*)d_in[7];
    const float* be1 = (const float*)d_in[8];
    const float* g2  = (const float*)d_in[9];
    const float* be2 = (const float*)d_in[10];
    const float* W1  = (const float*)d_in[11];
    const float* bb1 = (const float*)d_in[12];
    const float* W2  = (const float*)d_in[13];
    const float* bb2 = (const float*)d_in[14];
    float* out = (float*)d_out;

    constexpr int B = 2, T = 2048, D = 1024, H = 16;
    constexpr int BT = B * T, DFF = 4 * D;

    char* ws = (char*)d_ws;
    size_t off = 0;
    auto take = [&](size_t bytes) -> char* {
        char* p = ws + off;
        off += (bytes + 255) & ~(size_t)255;
        return p;
    };
    bf16_t* wqkvT  = (bf16_t*)take((size_t)3 * D * D * 2);
    bf16_t* w1T    = (bf16_t*)take((size_t)D * DFF * 2);
    bf16_t* w2T    = (bf16_t*)take((size_t)DFF * D * 2);
    float*  h      = (float*) take((size_t)BT * D * 4);
    bf16_t* hb     = (bf16_t*)take((size_t)BT * D * 2);
    bf16_t* qkvtmp = (bf16_t*)take((size_t)BT * 3 * D * 2);
    bf16_t* Qb     = (bf16_t*)take((size_t)BT * D * 2);
    bf16_t* Kb     = (bf16_t*)take((size_t)BT * D * 2);
    bf16_t* Vb     = (bf16_t*)take((size_t)BT * D * 2);
    float*  obuf   = (float*) take((size_t)BT * D * 4);
    float*  h2     = (float*) take((size_t)BT * D * 4);
    bf16_t* h2b    = (bf16_t*)take((size_t)BT * D * 2);
    bf16_t* ff1    = (bf16_t*)take((size_t)BT * DFF * 2);

    // 1) weight conversion
    cvt_wqkv_kernel<<<(3 * D * D + 255) / 256, 256, 0, stream>>>(Wq, Wk, Wv, wqkvT, D);
    transpose_cvt_kernel<<<((long)D * DFF + 255) / 256, 256, 0, stream>>>(W1, w1T, D, DFF);
    transpose_cvt_kernel<<<((long)DFF * D + 255) / 256, 256, 0, stream>>>(W2, w2T, DFF, D);
    // 2) LN1
    ln_kernel<<<BT, 256, 0, stream>>>(x, nullptr, g1, be1, h, hb, D);
    // 3) QKV projection (bf16 WMMA GEMM) + rearrange (Q pre-scaled by D^-0.5 = 1/32)
    gemm_bf16_kernel<<<dim3(3 * D / 256, BT / 64), 256, 0, stream>>>(
        hb, wqkvT, BT, 3 * D, D, nullptr, 0, nullptr, nullptr, qkvtmp);
    qkv_rearr_kernel<<<((long)BT * D + 255) / 256, 256, 0, stream>>>(
        qkvtmp, bq, bk, bv, Qb, Kb, Vb, T, H, 0.03125f);
    // 4) causal flash attention
    attn_kernel<<<dim3(T / 64, B * H), 128, 0, stream>>>(Qb, Kb, Vb, obuf, T, H, D);
    // 5) x2 = h + o; LN2
    ln_kernel<<<BT, 256, 0, stream>>>(h, obuf, g2, be2, h2, h2b, D);
    // 6) FFN: relu(h2 @ W1 + bb1) @ W2 + bb2 + h2
    gemm_bf16_kernel<<<dim3(DFF / 256, BT / 64), 256, 0, stream>>>(
        h2b, w1T, BT, DFF, D, bb1, 1, nullptr, nullptr, ff1);
    gemm_bf16_kernel<<<dim3(D / 256, BT / 64), 256, 0, stream>>>(
        ff1, w2T, BT, D, DFF, bb2, 0, h2, out, nullptr);
}
